// GalerkinAttention_65317862637749
// MI455X (gfx1250) — matrix operand order repaired and checked by reading, not compiled, not run on hardware
//
#include <hip/hip_runtime.h>
#include <hip/hip_bf16.h>

// ---------------------------------------------------------------------------
// Galerkin linear attention on MI455X (gfx1250, wave32, WMMA).
//
// All GEMMs: bf16 v_wmma_f32_16x16x32_bf16 with f32 accumulation.
// A one-time conversion pass produces bf16 operands; A-side tensors are
// stored plain, each aligned 32-element K-group in contiguous k order
// (k = 16*half + s, as on the B side), so every WMMA A fragment is a single
// contiguous 32B load (same k order as the B fragment), and every
// B fragment is naturally a contiguous 32B load. Inner loops are therefore
// pure vector-load + WMMA with no conversion/repacking VALU.
//
// Plain    (A-side): x, q/attn (in-place buffer), vT.
// Plain    (B-side): W_qkv, W_out, kT, pT.
// ---------------------------------------------------------------------------

typedef __attribute__((ext_vector_type(16))) __bf16          v16bf;
typedef __attribute__((ext_vector_type(16))) unsigned short  v16u;
typedef __attribute__((ext_vector_type(8)))  unsigned short  u16x8;
typedef __attribute__((ext_vector_type(8)))  float           f32x8;
typedef __attribute__((ext_vector_type(8)))  float           v8f;

__device__ __forceinline__ unsigned short f2bf(float f) {
  unsigned int u = __float_as_uint(f);
  u += 0x7fffu + ((u >> 16) & 1u);   // round-to-nearest-even
  return (unsigned short)(u >> 16);
}

// storage position of an offset within an aligned 32-element K-group:
// identity (contiguous k order, same as B side)
__device__ __forceinline__ int swz32(int o) {
  return o;
}

__device__ __forceinline__ u16x8 cvt8f(const float* p) {
  f32x8 v = *(const f32x8*)p;
  u16x8 r;
#pragma unroll
  for (int i = 0; i < 8; ++i) r[i] = f2bf(v[i]);
  return r;
}

// single 32B load of 16 bf16 (32B aligned)
__device__ __forceinline__ v16u ld16(const unsigned short* p) {
  return *(const v16u*)p;
}

__device__ __forceinline__ v8f wmma_bf16(v16u a, v16u b, v8f c) {
  union U { v16u u; v16bf h; } A, B;
  A.u = a; B.u = b;
  return __builtin_amdgcn_wmma_f32_16x16x32_bf16(false, A.h, false, B.h,
                                                 (short)0, c, false, false);
}

// ---------------------------------------------------------------------------
// K0: f32 -> bf16 conversion, one aligned 32-element group per thread.
// swz!=0 permutes each K-group to blocks [0..7, 16..23, 8..15, 24..31].
// ---------------------------------------------------------------------------
__global__ __launch_bounds__(256) void gal_cvt_bf16(
    const float* __restrict__ in, unsigned short* __restrict__ out,
    int n32, int swz)
{
  int g = blockIdx.x * blockDim.x + threadIdx.x;
  if (g >= n32) return;
  const float* p = in + (size_t)g * 32;
  unsigned short* q = out + (size_t)g * 32;
  u16x8 c0 = cvt8f(p);
  u16x8 c1 = cvt8f(p + 8);
  u16x8 c2 = cvt8f(p + 16);
  u16x8 c3 = cvt8f(p + 24);
  if (swz) { u16x8 t = c1; c1 = c2; c2 = t; }
  *(u16x8*)(q)      = c0;
  *(u16x8*)(q + 8)  = c1;
  *(u16x8*)(q + 16) = c2;
  *(u16x8*)(q + 24) = c3;
}

// ---------------------------------------------------------------------------
// Stage 1: qkv = x[32768,512] @ Wqkv[1536,512]^T, fused per-head LayerNorm.
// grid = (24, 64), block = 256 (8 waves x 64x64 tiles => 512 rows x 64 cols).
// blockIdx.x: 0-7 => q heads, 8-15 => k heads, 16-23 => v heads.
// ---------------------------------------------------------------------------
__global__ __launch_bounds__(256) void gal_qkv_ln(
    const unsigned short* __restrict__ xs,    // bf16, plain
    const unsigned short* __restrict__ Wb,    // bf16, plain
    const float* __restrict__ gK, const float* __restrict__ bK,
    const float* __restrict__ gV, const float* __restrict__ bV,
    unsigned short* __restrict__ qbuf,   // [32768,512] bf16, plain cols
    unsigned short* __restrict__ kT,     // [32,64,8192] bf16, plain tok
    unsigned short* __restrict__ vT)     // [32,64,8192] bf16, plain tok
{
  const int lane = threadIdx.x & 31;
  const int wave = threadIdx.x >> 5;
  const int hi   = lane >> 4;
  const int lm   = lane & 15;
  const int n0g  = blockIdx.x * 64;
  const int m0   = blockIdx.y * 512 + wave * 64;

  const unsigned short* pa[4];
  const unsigned short* pb[4];
#pragma unroll
  for (int t = 0; t < 4; ++t) {
    pa[t] = xs + (size_t)(m0  + t*16 + lm) * 512 + hi*16;
    pb[t] = Wb + (size_t)(n0g + t*16 + lm) * 512 + hi*16;
  }

  v8f acc[4][4] = {};
  for (int k0 = 0; k0 < 512; k0 += 32) {
    v16u a[4], b[4];
#pragma unroll
    for (int t = 0; t < 4; ++t) { a[t] = ld16(pa[t] + k0); b[t] = ld16(pb[t] + k0); }
#pragma unroll
    for (int tm = 0; tm < 4; ++tm)
#pragma unroll
      for (int tn = 0; tn < 4; ++tn)
        acc[tm][tn] = wmma_bf16(a[tm], b[tn], acc[tm][tn]);
  }

  const int sel  = blockIdx.x >> 3;  // 0=q 1=k 2=v
  const int h    = blockIdx.x & 7;
  const int b    = m0 >> 13;
  const int tok0 = m0 & 8191;

  if (sel == 0) {
    // store q with plain column positions (swz32 is identity) (q is stage-3 A operand)
    int colsw[4];
#pragma unroll
    for (int tn = 0; tn < 4; ++tn) {
      int o = tn*16 + lm;
      colsw[tn] = (o & ~31) | swz32(o & 31);
    }
#pragma unroll
    for (int tm = 0; tm < 4; ++tm)
#pragma unroll
      for (int r = 0; r < 8; ++r) {
        unsigned short* pr = qbuf + (size_t)(m0 + tm*16 + hi*8 + r) * 512 + h*64;
#pragma unroll
        for (int tn = 0; tn < 4; ++tn)
          pr[colsw[tn]] = f2bf(acc[tm][tn][r]);
      }
  } else {
    const float* gamma = (sel == 1 ? gK : gV) + h*64;
    const float* beta  = (sel == 1 ? bK : bV) + h*64;
    unsigned short* T  = (sel == 1 ? kT : vT) + (size_t)(b*8 + h) * 64 * 8192;
    float g[4], be[4];
#pragma unroll
    for (int tn = 0; tn < 4; ++tn) { g[tn] = gamma[tn*16 + lm]; be[tn] = beta[tn*16 + lm]; }
#pragma unroll
    for (int tm = 0; tm < 4; ++tm)
#pragma unroll
      for (int r = 0; r < 8; ++r) {
        // LayerNorm stats over 64 cols: 4 frag-cols/lane + 16-lane butterfly
        float s1 = 0.f, s2 = 0.f;
#pragma unroll
        for (int tn = 0; tn < 4; ++tn) { float v = acc[tm][tn][r]; s1 += v; s2 += v*v; }
#pragma unroll
        for (int mk = 1; mk <= 8; mk <<= 1) {
          s1 += __shfl_xor(s1, mk, 32);
          s2 += __shfl_xor(s2, mk, 32);
        }
        const float mean = s1 * (1.0f/64.0f);
        const float var  = s2 * (1.0f/64.0f) - mean * mean;
        const float rs   = rsqrtf(var + 1e-5f);
        const int   tok  = tok0 + tm*16 + hi*8 + r;
        // k stored plain (stage-2 B); v stored plain too (stage-2 A)
        const int tokidx = (sel == 1) ? tok : ((tok & ~31) | swz32(tok & 31));
#pragma unroll
        for (int tn = 0; tn < 4; ++tn) {
          float y = (acc[tm][tn][r] - mean) * rs * g[tn] + be[tn];
          T[(size_t)(tn*16 + lm) * 8192 + tokidx] = f2bf(y);
        }
      }
  }
}

// ---------------------------------------------------------------------------
// Stage 2: pT[bh][j][i] = (1/n) * sum_t v[t][j] * k[t][i]   (64x64, K=8192)
// grid = 32 blocks (one per (b,h)); wave w owns (tm=w&3, tn in {2*(w>>2)+0,1}).
// ---------------------------------------------------------------------------
__global__ __launch_bounds__(256) void gal_kv_outer(
    const unsigned short* __restrict__ kT,   // plain (B)
    const unsigned short* __restrict__ vT,   // plain (A)
    unsigned short* __restrict__ pT)         // [32,64,64] bf16 plain
{
  const int lane = threadIdx.x & 31, wave = threadIdx.x >> 5;
  const int hi = lane >> 4, lm = lane & 15;
  const int bh  = blockIdx.x;
  const int tm  = wave & 3;
  const int tn0 = (wave >> 2) * 2;
  const unsigned short* pa  = vT + (size_t)bh*64*8192 + (size_t)(tm*16 + lm)      * 8192 + hi*16;
  const unsigned short* pb0 = kT + (size_t)bh*64*8192 + (size_t)(tn0*16 + lm)     * 8192 + hi*16;
  const unsigned short* pb1 = kT + (size_t)bh*64*8192 + (size_t)((tn0+1)*16 + lm) * 8192 + hi*16;

  v8f acc0 = {}, acc1 = {};
  for (int k0 = 0; k0 < 8192; k0 += 32) {
    v16u a  = ld16(pa  + k0);
    v16u b0 = ld16(pb0 + k0);
    v16u b1 = ld16(pb1 + k0);
    acc0 = wmma_bf16(a, b0, acc0);
    acc1 = wmma_bf16(a, b1, acc1);
  }
  unsigned short* out = pT + (size_t)bh * 4096;
  const float sc = 1.0f / 8192.0f;   // divide by n
#pragma unroll
  for (int r = 0; r < 8; ++r) {
    unsigned short* pr = out + (size_t)(tm*16 + hi*8 + r) * 64;
    pr[tn0*16 + lm]     = f2bf(acc0[r] * sc);
    pr[(tn0+1)*16 + lm] = f2bf(acc1[r] * sc);
  }
}

// ---------------------------------------------------------------------------
// Stage 3: attn[tok, h*64+j] = sum_i q[tok, h*64+i] * pT[j][i]
// In place over the q buffer; reads plain q, writes plain attn cols
// (attn is stage-4's A operand). grid = (32 bh, 16), block = 256.
// ---------------------------------------------------------------------------
__global__ __launch_bounds__(256) void gal_q_p(
    const unsigned short* __restrict__ pT,
    unsigned short* __restrict__ qa)     // q in / attn out, [32768,512]
{
  const int lane = threadIdx.x & 31, wave = threadIdx.x >> 5;
  const int hi = lane >> 4, lm = lane & 15;
  const int bh = blockIdx.x;
  const int b = bh >> 3, h = bh & 7;
  const int m0 = blockIdx.y * 512 + wave * 64;
  unsigned short* qb = qa + ((size_t)b * 8192) * 512 + h * 64;
  const unsigned short* P = pT + (size_t)bh * 4096;

  v8f acc[4][4] = {};
#pragma unroll
  for (int k0 = 0; k0 < 64; k0 += 32) {
    v16u a[4], bm[4];
#pragma unroll
    for (int t = 0; t < 4; ++t) {
      a[t]  = ld16(qb + (size_t)(m0 + t*16 + lm) * 512 + k0 + hi*16);
      bm[t] = ld16(P  + (size_t)(t*16 + lm) * 64        + k0 + hi*16);
    }
#pragma unroll
    for (int tm = 0; tm < 4; ++tm)
#pragma unroll
      for (int tn = 0; tn < 4; ++tn)
        acc[tm][tn] = wmma_bf16(a[tm], bm[tn], acc[tm][tn]);
  }
  int colsw[4];
#pragma unroll
  for (int tn = 0; tn < 4; ++tn) {
    int o = tn*16 + lm;
    colsw[tn] = (o & ~31) | swz32(o & 31);
  }
#pragma unroll
  for (int tm = 0; tm < 4; ++tm)
#pragma unroll
    for (int r = 0; r < 8; ++r) {
      unsigned short* pr = qb + (size_t)(m0 + tm*16 + hi*8 + r) * 512;
#pragma unroll
      for (int tn = 0; tn < 4; ++tn)
        pr[colsw[tn]] = f2bf(acc[tm][tn][r]);
    }
}

// ---------------------------------------------------------------------------
// Stage 4: out = attn[32768,512] @ Wout[512,512]^T + b_out   (f32 output)
// grid = (8, 64), block = 256 (wave tile 64x64).
// ---------------------------------------------------------------------------
__global__ __launch_bounds__(256) void gal_out_proj(
    const unsigned short* __restrict__ attn,   // plain (A)
    const unsigned short* __restrict__ Wob,    // bf16 plain (B)
    const float* __restrict__ bout, float* __restrict__ out)
{
  const int lane = threadIdx.x & 31, wave = threadIdx.x >> 5;
  const int hi = lane >> 4, lm = lane & 15;
  const int n0 = blockIdx.x * 64;
  const int m0 = blockIdx.y * 512 + wave * 64;

  const unsigned short* pa[4];
  const unsigned short* pb[4];
#pragma unroll
  for (int t = 0; t < 4; ++t) {
    pa[t] = attn + (size_t)(m0 + t*16 + lm) * 512 + hi*16;
    pb[t] = Wob  + (size_t)(n0 + t*16 + lm) * 512 + hi*16;
  }

  v8f acc[4][4] = {};
  for (int k0 = 0; k0 < 512; k0 += 32) {
    v16u a[4], b[4];
#pragma unroll
    for (int t = 0; t < 4; ++t) { a[t] = ld16(pa[t] + k0); b[t] = ld16(pb[t] + k0); }
#pragma unroll
    for (int tm = 0; tm < 4; ++tm)
#pragma unroll
      for (int tn = 0; tn < 4; ++tn)
        acc[tm][tn] = wmma_bf16(a[tm], b[tn], acc[tm][tn]);
  }
  float bias[4];
#pragma unroll
  for (int tn = 0; tn < 4; ++tn) bias[tn] = bout[n0 + tn*16 + lm];
#pragma unroll
  for (int tm = 0; tm < 4; ++tm)
#pragma unroll
    for (int r = 0; r < 8; ++r) {
      float* pr = out + (size_t)(m0 + tm*16 + hi*8 + r) * 512 + n0;
#pragma unroll
      for (int tn = 0; tn < 4; ++tn)
        pr[tn*16 + lm] = acc[tm][tn][r] + bias[tn];   // coalesced f32 stores
    }
}

// ---------------------------------------------------------------------------
extern "C" void kernel_launch(void* const* d_in, const int* in_sizes, int n_in,
                              void* d_out, int out_size, void* d_ws, size_t ws_size,
                              hipStream_t stream) {
  const float* x    = (const float*)d_in[0];
  const float* Wqkv = (const float*)d_in[1];
  const float* gK   = (const float*)d_in[2];
  const float* bK   = (const float*)d_in[3];
  const float* gV   = (const float*)d_in[4];
  const float* bV   = (const float*)d_in[5];
  const float* Wout = (const float*)d_in[6];
  const float* bout = (const float*)d_in[7];
  float* out = (float*)d_out;

  // workspace layout (bf16 elements), ~137 MB total
  unsigned short* ws    = (unsigned short*)d_ws;
  unsigned short* xs    = ws;                                  // 32768*512
  unsigned short* Wqkvb = xs    + (size_t)32768 * 512;         // 1536*512
  unsigned short* Woutb = Wqkvb + (size_t)1536 * 512;          // 512*512
  unsigned short* qbuf  = Woutb + (size_t)512 * 512;           // 32768*512 (also attn)
  unsigned short* kT    = qbuf  + (size_t)32768 * 512;         // 32*64*8192
  unsigned short* vT    = kT    + (size_t)32 * 64 * 8192;
  unsigned short* pT    = vT    + (size_t)32 * 64 * 8192;      // 32*64*64

  // one-time bf16 conversion (all plain)
  hipLaunchKernelGGL(gal_cvt_bf16, dim3(2048), dim3(256), 0, stream,
                     x, xs, 32768 * 512 / 32, 0);
  hipLaunchKernelGGL(gal_cvt_bf16, dim3(96),   dim3(256), 0, stream,
                     Wqkv, Wqkvb, 1536 * 512 / 32, 0);
  hipLaunchKernelGGL(gal_cvt_bf16, dim3(32),   dim3(256), 0, stream,
                     Wout, Woutb, 512 * 512 / 32, 0);

  hipLaunchKernelGGL(gal_qkv_ln,   dim3(24, 64), dim3(256), 0, stream,
                     xs, Wqkvb, gK, bK, gV, bV, qbuf, kT, vT);
  hipLaunchKernelGGL(gal_kv_outer, dim3(32),     dim3(256), 0, stream, kT, vT, pT);
  hipLaunchKernelGGL(gal_q_p,      dim3(32, 16), dim3(256), 0, stream, pT, qbuf);
  hipLaunchKernelGGL(gal_out_proj, dim3(8, 64),  dim3(256), 0, stream,
                     qbuf, Woutb, bout, out);
}
